// CrossChannelAttention_77318001263034
// MI455X (gfx1250) — matrix operand order repaired and checked by reading, not compiled, not run on hardware
//
#include <hip/hip_runtime.h>
#include <hip/hip_bf16.h>
#include <stdint.h>

// ---------------- problem constants ----------------
#define Dm   192          // model dim
#define TD   576          // 3*D (qkv)
#define Ff   384          // ffn dim
#define MB   32           // tokens per workgroup (4 batches x S=8)
#define NTQ  36           // TD/16
#define NTO  12           // Dm/16
#define NTF1 24           // Ff/16
#define KS_D 6            // Dm/32 k-steps
#define KS_F 12           // Ff/32 k-steps

// packed bf16 weight offsets (in dwords) inside d_ws
#define OFF_QKV 0
#define OFF_OUT (OFF_QKV + KS_D*NTQ*32*8)   // 55296
#define OFF_F1  (OFF_OUT + KS_D*NTO*32*8)   // 73728
#define OFF_F2  (OFF_F1  + KS_D*NTF1*32*8)  // 110592  (end = 147456 dwords = 576KB)

typedef __attribute__((ext_vector_type(16))) __bf16 v16bf;
typedef __attribute__((ext_vector_type(8)))  float  v8f;

union Frag { uint32_t u[8]; v16bf v; };

__device__ __forceinline__ uint32_t f2bf1(float f) {   // fp32 -> bf16 (RNE)
    uint32_t u = __float_as_uint(f);
    return ((u + 0x7FFFu + ((u >> 16) & 1u)) >> 16) & 0xFFFFu;
}

// ---------------- weight pre-pack kernel ----------------
// Packs f32 W[K][N] into the CDNA5 WMMA bf16 B-fragment layout:
// dword idx = ((kt*NT + nt)*32 + lane)*8 + v holds bf16{W[k][n], W[k+1][n]}
// with n = nt*16 + lane%16, k = kt*32 + (v<4?0:16) + (lane<16?0:8) + 2(v&3).
__global__ void pack_b_bf16(const float* __restrict__ W, uint32_t* __restrict__ dst,
                            int K, int N) {
    int idx = blockIdx.x * blockDim.x + threadIdx.x;
    int ntiles = N >> 4;
    int total  = (K >> 5) * ntiles * 256;
    if (idx >= total) return;
    int v    = idx & 7;
    int lane = (idx >> 3) & 31;
    int t    = idx >> 8;
    int nt   = t % ntiles;
    int kt   = t / ntiles;
    int n = nt*16 + (lane & 15);
    int k = kt*32 + ((v < 4) ? 0 : 16) + ((lane < 16) ? 0 : 8) + 2*(v & 3);
    uint32_t lo = f2bf1(W[k*N + n]);
    uint32_t hi = f2bf1(W[(k+1)*N + n]);
    dst[idx] = lo | (hi << 16);
}

// ---------------- WMMA GEMM micro-block ----------------
// A: bf16 in LDS, row-major [M][Acols]; B: packed fragments; acc: NCH 16x16 f32 tiles.
template<int KSTEPS, int NCH>
__device__ __forceinline__ void wmma_block(const uint32_t* __restrict__ Adw, int Acols,
                                           int mtile, const uint32_t* __restrict__ bbase,
                                           int ntotal, int nt0, int lane, v8f* acc)
{
    const int m  = mtile*16 + (lane & 15);
    const int kb = (lane < 16) ? 0 : 8;
    for (int ks = 0; ks < KSTEPS; ++ks) {
        Frag a;
        const int k0 = ks*32;
        #pragma unroll
        for (int v = 0; v < 4; ++v) a.u[v]   = Adw[(m*Acols + k0 + kb      + 2*v) >> 1];
        #pragma unroll
        for (int v = 0; v < 4; ++v) a.u[4+v] = Adw[(m*Acols + k0 + 16 + kb + 2*v) >> 1];
        #pragma unroll
        for (int nt = 0; nt < NCH; ++nt) {
            Frag b;
            const uint32_t* bp = bbase + ((((ks*ntotal + nt0 + nt)*32) + lane) << 3);
            #pragma unroll
            for (int v = 0; v < 8; ++v) b.u[v] = bp[v];
            acc[nt] = __builtin_amdgcn_wmma_f32_16x16x32_bf16(
                          false, a.v, false, b.v, (short)0, acc[nt], false, false);
        }
    }
}

// ---------------- parallel LayerNorm: 4 lanes per row, wave32 butterfly ----------------
// 128 threads = 32 rows x 4 lanes; each lane owns a 48-element quarter.
__device__ __forceinline__ void layernorm_rows(const float* __restrict__ xsrc,
                                               unsigned short* __restrict__ dst,
                                               const float* __restrict__ g,
                                               const float* __restrict__ b, int tid)
{
    const int row = tid >> 2;          // 0..31
    const int q   = (tid & 3) * 48;    // quarter start
    const float* xr = xsrc + row*Dm;

    float s = 0.f;
    #pragma unroll 4
    for (int d = 0; d < 48; ++d) s += xr[q + d];
    s += __shfl_xor(s, 1, 32);
    s += __shfl_xor(s, 2, 32);
    const float mu = s * (1.0f/Dm);

    float var = 0.f;
    #pragma unroll 4
    for (int d = 0; d < 48; ++d) { float t = xr[q + d] - mu; var += t*t; }
    var += __shfl_xor(var, 1, 32);
    var += __shfl_xor(var, 2, 32);
    const float inv = rsqrtf(var*(1.0f/Dm) + 1e-5f);

    #pragma unroll 4
    for (int d = 0; d < 48; ++d) {
        const int c = q + d;
        float v = (xr[c] - mu) * inv * g[c] + b[c];
        dst[row*Dm + c] = (unsigned short)f2bf1(v);
    }
}

// ---------------- fused transformer-block kernel ----------------
__global__ __launch_bounds__(128, 1)
void fused_block_kernel(const float* __restrict__ x, const uint32_t* __restrict__ wpack,
                        const float* __restrict__ b_out,
                        const float* __restrict__ g1, const float* __restrict__ b1,
                        const float* __restrict__ g2, const float* __restrict__ b2,
                        const float* __restrict__ b_f1, const float* __restrict__ b_f2,
                        float* __restrict__ out)
{
    __shared__ float          xs[MB*Dm];    // 24 KB : x tile, later x2 (post-attn residual)
    __shared__ unsigned short lnb[MB*Dm];   // 12 KB : ln1 bf16 -> attn-out bf16 -> ln2 bf16
    __shared__ float          qkv[MB*TD];   // 72 KB : qkv f32, later h1 bf16

    const int  tid  = threadIdx.x;
    const int  lane = tid & 31;
    const int  wave = tid >> 5;
    const int  mi   = wave & 1;     // M-tile (rows 0-15 / 16-31)
    const int  ng   = wave >> 1;    // N-group
    const long base = (long)blockIdx.x * (MB*Dm);
    const uint32_t* Adw = (const uint32_t*)lnb;

    // ---- phase 0: load x tile ----
    for (int i = tid; i < MB*Dm; i += 128) xs[i] = x[base + i];
    __syncthreads();

    // ---- phase 0b: LayerNorm 1 -> lnb (bf16) ----
    layernorm_rows(xs, lnb, g1, b1, tid);
    __syncthreads();

    // ---- phase 1: qkv = ln1 @ w_qkv  (WMMA) ----
    for (int chunk = 0; chunk < 3; ++chunk) {
        v8f acc[6];
        #pragma unroll
        for (int i = 0; i < 6; ++i) acc[i] = (v8f){0.f,0.f,0.f,0.f,0.f,0.f,0.f,0.f};
        const int nt0 = ng*18 + chunk*6;
        wmma_block<KS_D,6>(Adw, Dm, mi, wpack + OFF_QKV, NTQ, nt0, lane, acc);
        #pragma unroll
        for (int nt = 0; nt < 6; ++nt) {
            const int c = (nt0+nt)*16 + (lane & 15);
            #pragma unroll
            for (int v = 0; v < 8; ++v) {
                const int r = mi*16 + v + ((lane < 16) ? 0 : 8);
                qkv[r*TD + c] = acc[nt][v];
            }
        }
    }
    __syncthreads();

    // ---- phase 2: attention (S=8, H=4, DH=48) -> lnb (bf16) ----
    {
        const int b  = tid >> 5;          // batch within tile (0..3)
        const int h  = (tid >> 3) & 3;    // head
        const int qr = tid & 7;           // query row
        const int rq = b*8 + qr;
        const float scale = 0.14433756729740643f;   // 1/sqrt(48)
        float sc[8];
        float mx = -1e30f;
        for (int kr = 0; kr < 8; ++kr) {
            float s = 0.f;
            const float* qp = &qkv[rq*TD + h*48];
            const float* kp = &qkv[(b*8+kr)*TD + 192 + h*48];
            #pragma unroll 4
            for (int d = 0; d < 48; ++d) s += qp[d]*kp[d];
            s *= scale; sc[kr] = s; mx = fmaxf(mx, s);
        }
        float sum = 0.f;
        for (int kr = 0; kr < 8; ++kr) { sc[kr] = __expf(sc[kr] - mx); sum += sc[kr]; }
        const float rs = 1.0f / sum;
        for (int d = 0; d < 48; ++d) {
            float o = 0.f;
            #pragma unroll
            for (int kr = 0; kr < 8; ++kr) o += sc[kr]*qkv[(b*8+kr)*TD + 384 + h*48 + d];
            lnb[rq*Dm + h*48 + d] = (unsigned short)f2bf1(o * rs);
        }
    }
    __syncthreads();

    // ---- phase 3: out-proj + bias + residual -> xs (x2) ----
    {
        v8f acc[6];
        #pragma unroll
        for (int i = 0; i < 6; ++i) acc[i] = (v8f){0.f,0.f,0.f,0.f,0.f,0.f,0.f,0.f};
        const int nt0 = ng*6;
        wmma_block<KS_D,6>(Adw, Dm, mi, wpack + OFF_OUT, NTO, nt0, lane, acc);
        #pragma unroll
        for (int nt = 0; nt < 6; ++nt) {
            const int c = (nt0+nt)*16 + (lane & 15);
            #pragma unroll
            for (int v = 0; v < 8; ++v) {
                const int r = mi*16 + v + ((lane < 16) ? 0 : 8);
                xs[r*Dm + c] = acc[nt][v] + b_out[c] + xs[r*Dm + c];
            }
        }
    }
    __syncthreads();

    // ---- phase 4: LayerNorm 2 -> lnb (bf16) ----
    layernorm_rows(xs, lnb, g2, b2, tid);
    __syncthreads();

    // ---- phase 5: FFN1 + exact GELU -> h1 (bf16, reuses qkv LDS) ----
    unsigned short* h1 = (unsigned short*)qkv;
    for (int chunk = 0; chunk < 2; ++chunk) {
        v8f acc[6];
        #pragma unroll
        for (int i = 0; i < 6; ++i) acc[i] = (v8f){0.f,0.f,0.f,0.f,0.f,0.f,0.f,0.f};
        const int nt0 = ng*12 + chunk*6;
        wmma_block<KS_D,6>(Adw, Dm, mi, wpack + OFF_F1, NTF1, nt0, lane, acc);
        #pragma unroll
        for (int nt = 0; nt < 6; ++nt) {
            const int c = (nt0+nt)*16 + (lane & 15);
            #pragma unroll
            for (int v = 0; v < 8; ++v) {
                const int r = mi*16 + v + ((lane < 16) ? 0 : 8);
                float val = acc[nt][v] + b_f1[c];
                val = 0.5f * val * (1.0f + erff(val * 0.70710678118654752f));
                h1[r*Ff + c] = (unsigned short)f2bf1(val);
            }
        }
    }
    __syncthreads();

    // ---- phase 6: FFN2 + bias + residual -> global out ----
    {
        v8f acc[6];
        #pragma unroll
        for (int i = 0; i < 6; ++i) acc[i] = (v8f){0.f,0.f,0.f,0.f,0.f,0.f,0.f,0.f};
        const int nt0 = ng*6;
        wmma_block<KS_F,6>((const uint32_t*)h1, Ff, mi, wpack + OFF_F2, NTO, nt0, lane, acc);
        #pragma unroll
        for (int nt = 0; nt < 6; ++nt) {
            const int c = (nt0+nt)*16 + (lane & 15);
            #pragma unroll
            for (int v = 0; v < 8; ++v) {
                const int r = mi*16 + v + ((lane < 16) ? 0 : 8);
                out[base + r*Dm + c] = acc[nt][v] + b_f2[c] + xs[r*Dm + c];
            }
        }
    }
}

// ---------------- host launcher ----------------
extern "C" void kernel_launch(void* const* d_in, const int* in_sizes, int n_in,
                              void* d_out, int out_size, void* d_ws, size_t ws_size,
                              hipStream_t stream) {
    const float* x     = (const float*)d_in[0];
    const float* w_qkv = (const float*)d_in[1];
    const float* w_out = (const float*)d_in[2];
    const float* b_out = (const float*)d_in[3];
    const float* g1    = (const float*)d_in[4];
    const float* b1    = (const float*)d_in[5];
    const float* g2    = (const float*)d_in[6];
    const float* b2    = (const float*)d_in[7];
    const float* w_f1  = (const float*)d_in[8];
    const float* b_f1  = (const float*)d_in[9];
    const float* w_f2  = (const float*)d_in[10];
    const float* b_f2  = (const float*)d_in[11];
    float*    out = (float*)d_out;
    uint32_t* wp  = (uint32_t*)d_ws;

    // pack weights into WMMA bf16 B-fragment layout (L2-resident, re-done every launch)
    {
        int t;
        t = (192/32)*(576/16)*256;
        pack_b_bf16<<<(t+255)/256, 256, 0, stream>>>(w_qkv, wp + OFF_QKV, 192, 576);
        t = (192/32)*(192/16)*256;
        pack_b_bf16<<<(t+255)/256, 256, 0, stream>>>(w_out, wp + OFF_OUT, 192, 192);
        t = (192/32)*(384/16)*256;
        pack_b_bf16<<<(t+255)/256, 256, 0, stream>>>(w_f1,  wp + OFF_F1,  192, 384);
        t = (384/32)*(192/16)*256;
        pack_b_bf16<<<(t+255)/256, 256, 0, stream>>>(w_f2,  wp + OFF_F2,  384, 192);
    }

    const int tokens = in_sizes[0] / Dm;   // B*S = 262144
    const int blocks = tokens / MB;        // 8192
    fused_block_kernel<<<blocks, 128, 0, stream>>>(x, wp, b_out, g1, b1, g2, b2,
                                                   b_f1, b_f2, out);
}
